// AdaptiveHalting_16793322127879
// MI455X (gfx1250) — matrix operand order repaired and checked by reading, not compiled, not run on hardware
//
#include <hip/hip_runtime.h>
#include <hip/hip_bf16.h>

typedef __attribute__((ext_vector_type(16))) __bf16 v16bf;
typedef __attribute__((ext_vector_type(8)))  float  v8f;
typedef __attribute__((ext_vector_type(4)))  int    v4i;

#define B_SZ   8192
#define H_SZ   2048
#define HALF_SZ 1024
#define MAX_STEPS 8

// CDNA5 async global->LDS path (ASYNCcnt-tracked), with safe fallback.
#if __has_builtin(__builtin_amdgcn_global_load_async_to_lds_b128)
#define HAVE_ASYNC_LDS 1
#else
#define HAVE_ASYNC_LDS 0
#endif

// Pointer types matching the builtin's signature: (int4 AS1*, int4 AS3*, imm, imm)
typedef __attribute__((address_space(1))) v4i* gptr_v4i;
typedef __attribute__((address_space(3))) v4i* lptr_v4i;

__device__ __forceinline__ void copy16_to_lds(__bf16* l, const __bf16* g)
{
#if HAVE_ASYNC_LDS
  __builtin_amdgcn_global_load_async_to_lds_b128(
      (gptr_v4i)(uintptr_t)g, (lptr_v4i)(uintptr_t)l, 0, 0);
#else
  *(uint4*)l = *(const uint4*)g;
#endif
}

__device__ __forceinline__ void wait_stage_done()
{
#if HAVE_ASYNC_LDS
  asm volatile("s_wait_asynccnt 0" ::: "memory");
#endif
  __syncthreads();
}

// ---------------- bf16 WMMA GEMM: C[M][N] = A[M][K] * Bt[N][K]^T + bias ----------------
// Block tile 128x256, BK=32, 8 waves as 2(M) x 4(N), each wave computes 64x64
// (16 v_wmma_f32_16x16x32_bf16 per k-step). Double-buffered LDS, async staging.
constexpr int BM = 128, BN = 256, BK = 32;
constexpr int LDT = 40;  // LDS row stride in bf16 elems: 80B -> 16B aligned, conflict-free

template<bool RELU>
__global__ __launch_bounds__(256) void gemm_bf16_wmma(
    const __bf16* __restrict__ A,    // [M][K] row-major bf16
    const __bf16* __restrict__ Bt,   // [N][K] row-major bf16 (transposed weight)
    const float*  __restrict__ bias, // [N]
    float* __restrict__ C,           // [M][N] f32
    const int* __restrict__ stop_flag,
    int M, int N, int K)
{
  if (*stop_flag) return;   // uniform early-exit: reproduces the data-dependent break

  __shared__ __bf16 As[2][BM * LDT];
  __shared__ __bf16 Bs[2][BN * LDT];

  const int tid  = threadIdx.x;
  const int wave = tid >> 5;
  const int lane = tid & 31;
  const int wm   = wave & 1;        // 2 waves over M (64 rows each)
  const int wn   = wave >> 1;       // 4 waves over N (64 cols each)
  const int lr   = lane & 15;       // row/col within 16x16 fragment
  const int lh   = lane >> 4;       // K-half selector per ISA A/B layout

  const int m0 = blockIdx.y * BM;
  const int n0 = blockIdx.x * BN;

  const __bf16* Ap = A  + (size_t)m0 * K;
  const __bf16* Bp = Bt + (size_t)n0 * K;

  // Stage one 16B chunk per (thread, iter): A = 512 chunks, B = 1024 chunks.
  auto stage = [&](int buf, int k0) {
    #pragma unroll
    for (int i = 0; i < 2; ++i) {
      int c = tid + i * 256, row = c >> 2, quad = c & 3;
      copy16_to_lds(&As[buf][row * LDT + quad * 8],
                    Ap + (size_t)row * K + k0 + quad * 8);
    }
    #pragma unroll
    for (int i = 0; i < 4; ++i) {
      int c = tid + i * 256, row = c >> 2, quad = c & 3;
      copy16_to_lds(&Bs[buf][row * LDT + quad * 8],
                    Bp + (size_t)row * K + k0 + quad * 8);
    }
  };

  v8f acc[4][4];
  #pragma unroll
  for (int s = 0; s < 4; ++s)
    #pragma unroll
    for (int t = 0; t < 4; ++t)
      acc[s][t] = (v8f){0.f,0.f,0.f,0.f,0.f,0.f,0.f,0.f};

  const int nk = K / BK;
  stage(0, 0);
  wait_stage_done();

  for (int kt = 0; kt < nk; ++kt) {
    const int buf = kt & 1;
    if (kt + 1 < nk) stage(buf ^ 1, (kt + 1) * BK);   // prefetch into other buffer

    // A fragments: lane lr = row, lh selects K {0..7,16..23} vs {8..15,24..31}
    v16bf af[4];
    #pragma unroll
    for (int s = 0; s < 4; ++s) {
      const __bf16* p = &As[buf][(wm * 64 + s * 16 + lr) * LDT + lh * 8];
      union { v16bf v; uint4 q[2]; } u;
      u.q[0] = *(const uint4*)p;          // K base .. base+7
      u.q[1] = *(const uint4*)(p + 16);   // K base+16 .. base+23
      af[s] = u.v;
    }
    // B fragments: lane lr = column (row of Bt), lh selects K {0..7,16..23} vs {8..15,24..31}
    v16bf bfr[4];
    #pragma unroll
    for (int t = 0; t < 4; ++t) {
      const __bf16* p = &Bs[buf][(wn * 64 + t * 16 + lr) * LDT + lh * 8];
      union { v16bf v; uint4 q[2]; } u;
      u.q[0] = *(const uint4*)p;          // K base .. base+7
      u.q[1] = *(const uint4*)(p + 16);   // K base+16 .. base+23
      bfr[t] = u.v;
    }

    #pragma unroll
    for (int s = 0; s < 4; ++s)
      #pragma unroll
      for (int t = 0; t < 4; ++t)
        acc[s][t] = __builtin_amdgcn_wmma_f32_16x16x32_bf16(
            false, af[s], false, bfr[t], (short)0, acc[s][t], false, false);

    wait_stage_done();
  }

  // Epilogue: C/D layout -> VGPR v holds M = v + 8*lh, N = lr
  #pragma unroll
  for (int s = 0; s < 4; ++s) {
    #pragma unroll
    for (int t = 0; t < 4; ++t) {
      int n   = n0 + wn * 64 + t * 16 + lr;
      float b = bias[n];
      #pragma unroll
      for (int v = 0; v < 8; ++v) {
        int m = m0 + wm * 64 + s * 16 + lh * 8 + v;
        float val = acc[s][t][v] + b;
        if (RELU) val = fmaxf(val, 0.f);
        C[(size_t)m * N + n] = val;
      }
    }
  }
}

// ---------------- fp32 -> bf16 activation prep (optionally fused add) ----------------
__global__ __launch_bounds__(256) void to_bf16_kernel(
    const float* __restrict__ a, const float* __restrict__ b,
    __bf16* __restrict__ out, const int* __restrict__ stop_flag)
{
  if (*stop_flag) return;
  size_t i = ((size_t)blockIdx.x * 256 + threadIdx.x) * 4;
  float4 va = *(const float4*)(a + i);
  if (b) {
    float4 vb = *(const float4*)(b + i);
    va.x += vb.x; va.y += vb.y; va.z += vb.z; va.w += vb.w;
  }
  union { __bf16 h[4]; uint2 u; } o;
  o.h[0] = (__bf16)va.x; o.h[1] = (__bf16)va.y;
  o.h[2] = (__bf16)va.z; o.h[3] = (__bf16)va.w;
  *(uint2*)(out + i) = o.u;
}

// ---------------- weight convert + transpose: W[K][N] f32 -> Wt[N][K] bf16 ----------------
__global__ __launch_bounds__(256) void weight_bf16_t(
    const float* __restrict__ w, __bf16* __restrict__ o, int K, int N)
{
  size_t idx = (size_t)blockIdx.x * 256 + threadIdx.x;
  if (idx >= (size_t)K * N) return;
  int k = (int)(idx / N), n = (int)(idx % N);
  o[(size_t)n * K + k] = (__bf16)w[idx];
}

// ---------------- halt head: p = sigmoid(h . hw2 + hb2); ACT update ----------------
__global__ __launch_bounds__(256) void halt_update(
    const float* __restrict__ h,        // [B][HALF]
    const float* __restrict__ hw2,      // [HALF]
    const float* __restrict__ hb2,      // [1]
    const float* __restrict__ state,    // [B][H]
    float* __restrict__ acc,            // [B][H] == d_out
    float* __restrict__ remaining,      // [B]
    unsigned* __restrict__ rmax,        // this step's max-slot (float bits, >=0)
    const int* __restrict__ stop_flag,
    int last_step)
{
  if (*stop_flag) return;
  const int wave = threadIdx.x >> 5, lane = threadIdx.x & 31;
  const int row  = blockIdx.x * 8 + wave;

  float rem = remaining[row];
  float hw;
  if (last_step) {
    hw = rem;
  } else {
    float sum = 0.f;
    const float* hp = h + (size_t)row * HALF_SZ;
    for (int c = lane; c < HALF_SZ; c += 32) sum += hp[c] * hw2[c];
    #pragma unroll
    for (int off = 16; off; off >>= 1) sum += __shfl_down(sum, off, 32);
    sum = __shfl(sum, 0, 32);
    float p = 1.f / (1.f + __expf(-(sum + hb2[0])));
    hw = p * rem;
  }

  const float* sp = state + (size_t)row * H_SZ;
  float* ap = acc + (size_t)row * H_SZ;
  for (int c = lane; c < H_SZ; c += 32) ap[c] += hw * sp[c];

  float newrem = rem - hw;
  if (lane == 0) remaining[row] = newrem;

  __shared__ float wmax[8];
  if (lane == 0) wmax[wave] = newrem;
  __syncthreads();
  if (threadIdx.x == 0) {
    float m = wmax[0];
    #pragma unroll
    for (int i = 1; i < 8; ++i) m = fmaxf(m, wmax[i]);
    atomicMax(rmax, __float_as_uint(m));   // remaining >= 0 -> uint order == float order
  }
}

__global__ void step_flag(const unsigned* __restrict__ rmax, int* __restrict__ stop_flag)
{
  if (__uint_as_float(*rmax) < 0.5f) *stop_flag = 1;
}

// ---------------- LayerNorm + ReLU -> bf16, one wave per row ----------------
__global__ __launch_bounds__(256) void ln_relu_bf16(
    const float* __restrict__ y, const float* __restrict__ g, const float* __restrict__ b,
    __bf16* __restrict__ out, const int* __restrict__ stop_flag)
{
  if (*stop_flag) return;
  const int wave = threadIdx.x >> 5, lane = threadIdx.x & 31;
  const int row  = blockIdx.x * 8 + wave;
  const float* yp = y + (size_t)row * H_SZ;

  float s = 0.f, s2 = 0.f;
  for (int c = lane; c < H_SZ; c += 32) { float v = yp[c]; s += v; s2 += v * v; }
  #pragma unroll
  for (int off = 16; off; off >>= 1) {
    s  += __shfl_down(s,  off, 32);
    s2 += __shfl_down(s2, off, 32);
  }
  s  = __shfl(s,  0, 32);
  s2 = __shfl(s2, 0, 32);
  float mu   = s * (1.f / H_SZ);
  float var  = s2 * (1.f / H_SZ) - mu * mu;
  float rinv = rsqrtf(var + 1e-5f);

  __bf16* op = out + (size_t)row * H_SZ;
  for (int c = lane; c < H_SZ; c += 32) {
    float v = fmaxf((yp[c] - mu) * rinv * g[c] + b[c], 0.f);
    op[c] = (__bf16)v;
  }
}

__global__ void init_remaining(float* __restrict__ rem)
{
  rem[blockIdx.x * 256 + threadIdx.x] = 1.0f;
}

// ---------------- host-side orchestration ----------------
extern "C" void kernel_launch(void* const* d_in, const int* in_sizes, int n_in,
                              void* d_out, int out_size, void* d_ws, size_t ws_size,
                              hipStream_t stream)
{
  const float* initial_state = (const float*)d_in[0];
  const float* input_signal  = (const float*)d_in[1];
  const float* hw1 = (const float*)d_in[2];
  const float* hb1 = (const float*)d_in[3];
  const float* hw2 = (const float*)d_in[4];
  const float* hb2 = (const float*)d_in[5];
  const float* tw1 = (const float*)d_in[6];
  const float* tb1 = (const float*)d_in[7];
  const float* ln_g = (const float*)d_in[8];
  const float* ln_b = (const float*)d_in[9];
  const float* tw2 = (const float*)d_in[10];
  const float* tb2 = (const float*)d_in[11];
  float* acc = (float*)d_out;

  char* ws = (char*)d_ws;
  float*  state   = (float*)ws;   ws += (size_t)B_SZ * H_SZ * 4;
  __bf16* act_bf  = (__bf16*)ws;  ws += (size_t)B_SZ * H_SZ * 2;
  float*  hbuf    = (float*)ws;   ws += (size_t)B_SZ * HALF_SZ * 4;
  float*  ybuf    = (float*)ws;   ws += (size_t)B_SZ * H_SZ * 4;
  __bf16* g_bf    = (__bf16*)ws;  ws += (size_t)B_SZ * H_SZ * 2;
  __bf16* hw1_bf  = (__bf16*)ws;  ws += (size_t)HALF_SZ * H_SZ * 2;   // [1024][2048]
  __bf16* tw1_bf  = (__bf16*)ws;  ws += (size_t)H_SZ * H_SZ * 2;      // [2048][2048]
  __bf16* tw2_bf  = (__bf16*)ws;  ws += (size_t)H_SZ * H_SZ * 2;
  float*  remaining = (float*)ws; ws += (size_t)B_SZ * 4;
  unsigned* rmax  = (unsigned*)ws; ws += MAX_STEPS * 4;
  int* stop_flag  = (int*)ws;      ws += 4;

  // init (all async, graph-capture safe)
  (void)hipMemcpyAsync(state, initial_state, (size_t)B_SZ * H_SZ * 4,
                       hipMemcpyDeviceToDevice, stream);
  (void)hipMemsetAsync(acc, 0, (size_t)out_size * sizeof(float), stream);
  (void)hipMemsetAsync(rmax, 0, MAX_STEPS * 4 + 4, stream);   // rmax slots + stop_flag
  init_remaining<<<B_SZ / 256, 256, 0, stream>>>(remaining);

  weight_bf16_t<<<(H_SZ * HALF_SZ + 255) / 256, 256, 0, stream>>>(hw1, hw1_bf, H_SZ, HALF_SZ);
  weight_bf16_t<<<(H_SZ * H_SZ + 255) / 256, 256, 0, stream>>>(tw1, tw1_bf, H_SZ, H_SZ);
  weight_bf16_t<<<(H_SZ * H_SZ + 255) / 256, 256, 0, stream>>>(tw2, tw2_bf, H_SZ, H_SZ);

  const int nconv_blocks = (B_SZ * H_SZ) / (256 * 4);
  const dim3 gemm_blk(256);
  const dim3 grid_halt(HALF_SZ / BN, B_SZ / BM);   // 4 x 64
  const dim3 grid_full(H_SZ / BN, B_SZ / BM);      // 8 x 64

  for (int s = 0; s < MAX_STEPS; ++s) {
    if (s < MAX_STEPS - 1) {
      // halt hidden: h = relu(state @ hw1 + hb1)   (p unused on the last step)
      to_bf16_kernel<<<nconv_blocks, 256, 0, stream>>>(state, nullptr, act_bf, stop_flag);
      gemm_bf16_wmma<true><<<grid_halt, gemm_blk, 0, stream>>>(
          act_bf, hw1_bf, hb1, hbuf, stop_flag, B_SZ, HALF_SZ, H_SZ);
    }
    // p, ACT accumulation, remaining, global max
    halt_update<<<B_SZ / 8, 256, 0, stream>>>(
        hbuf, hw2, hb2, state, acc, remaining, rmax + s, stop_flag,
        (s == MAX_STEPS - 1) ? 1 : 0);

    if (s < MAX_STEPS - 1) {
      step_flag<<<1, 1, 0, stream>>>(rmax + s, stop_flag);
      // transition: state = relu(LN((state+input) @ tw1 + tb1)) @ tw2 + tb2
      to_bf16_kernel<<<nconv_blocks, 256, 0, stream>>>(state, input_signal, act_bf, stop_flag);
      gemm_bf16_wmma<false><<<grid_full, gemm_blk, 0, stream>>>(
          act_bf, tw1_bf, tb1, ybuf, stop_flag, B_SZ, H_SZ, H_SZ);
      ln_relu_bf16<<<B_SZ / 8, 256, 0, stream>>>(ybuf, ln_g, ln_b, g_bf, stop_flag);
      gemm_bf16_wmma<false><<<grid_full, gemm_blk, 0, stream>>>(
          g_bf, tw2_bf, tb2, state, stop_flag, B_SZ, H_SZ, H_SZ);
    }
  }
}
